// EncoderDecoder_77988016161282
// MI455X (gfx1250) — matrix operand order repaired and checked by reading, not compiled, not run on hardware
//
#include <hip/hip_runtime.h>
#include <math.h>

// ---------------- problem constants ----------------
#define B_   64
#define T_   128
#define I_   128
#define H_   512
#define FH_  2048   // 4*H
#define V_   32
#define L_   64
#define NWG  64     // persistent grid: 2 dirs x 32 N-blocks (64 cols each)

#define LDS_ROW_H   520      // 512 halves + 4-dword TDM pad -> 1040B row stride (bank-conflict free)
#define LDS_BYTES   (64 * LDS_ROW_H * 2)   // 66560 B dynamic LDS

typedef __attribute__((ext_vector_type(16))) _Float16 v16h;
typedef __attribute__((ext_vector_type(8)))  _Float16 v8h;
typedef __attribute__((ext_vector_type(8)))  float    v8f;
typedef __attribute__((ext_vector_type(4)))  unsigned v4u;
typedef __attribute__((ext_vector_type(8)))  int      v8i;
typedef __attribute__((ext_vector_type(4)))  int      v4i;

// ---------------- workspace layout (bytes) ----------------
constexpr size_t OFF_CNT  = 0;                        // 4   (barrier arrivals, monotonic)
constexpr size_t OFF_GEN  = 128;                      // 4   (barrier generation, monotonic)
constexpr size_t OFF_TOK  = 256;                      // 64*4 decoder token ids
constexpr size_t OFF_H16  = 512;                      // (2,64,512) f16 hidden
constexpr size_t OFF_CB   = OFF_H16 + 131072;         // (2,64,512) f32 cell
constexpr size_t OFF_GB   = OFF_CB  + 262144;         // (2,64,2048) f32 gates
constexpr size_t OFF_LB   = OFF_GB  + 1048576;        // (64,32) f32 logits
constexpr size_t OFF_BSE  = OFF_LB  + 8192;           // (2,2048) f32 enc bias sum
constexpr size_t OFF_BSD  = OFF_BSE + 16384;          // (2,2048) f32 dec bias sum
constexpr size_t OFF_EWHH = OFF_BSD + 16384;          // (2,2048,512) f16
constexpr size_t OFF_DWHH = OFF_EWHH + 4194304;       // (2,2048,512) f16
constexpr size_t OFF_EWIH = OFF_DWHH + 4194304;       // (2,2048,128) f16
constexpr size_t OFF_SIG  = OFF_EWIH + 1048576;       // (64,128,128) f16
constexpr size_t WS_NEED  = OFF_SIG + 2097152;        // ~12.9 MB total

__device__ __forceinline__ float sigm(float x) { return 1.0f / (1.0f + __expf(-x)); }

// Monotonic-ticket grid barrier (deterministic, graph-capture safe).
__device__ __forceinline__ void grid_sync(unsigned* cnt, unsigned* gen, unsigned round) {
    __syncthreads();
    if (threadIdx.x == 0) {
        __threadfence();
        unsigned arrived = __atomic_add_fetch(cnt, 1u, __ATOMIC_ACQ_REL);
        if (arrived == (unsigned)NWG * round) {
            __atomic_store_n(gen, round, __ATOMIC_RELEASE);
        } else {
            while (__atomic_load_n(gen, __ATOMIC_ACQUIRE) < round) {
                __builtin_amdgcn_s_sleep(1);
            }
        }
        __threadfence();
    }
    __syncthreads();
}

// TDM stage: DMA this direction's h panel (64 x 512 f16, row-padded by 16B) into LDS.
// Wave 0 issues one tensor_load_to_lds (TDM ignores EXEC; branch avoids duplicates),
// waits on TENSORcnt, then the block barrier releases the other waves.
__device__ __forceinline__ void stage_h(v4u g0, v8i g1) {
    if (threadIdx.x < 32) {
        v4i z4 = {0, 0, 0, 0};
        v8i z8 = {0, 0, 0, 0, 0, 0, 0, 0};
        __builtin_amdgcn_tensor_load_to_lds(g0, g1, z4, z4, z8, 0);
        __builtin_amdgcn_s_wait_tensorcnt(0);
    }
    __syncthreads();
}

// One recurrent-step gate GEMM: gbuf[d, 0:64, 0:2048] = h @ Whh^T (+ x_t @ Wih^T).
// A comes from the LDS-staged h panel (ds_load_b128), B streams from L2 (global_load_b128).
// Per wave: two 16x16 f32 accumulator tiles, K in chunks of 32 via wmma_f32_16x16x32_f16.
// A-frag  : lane L row m=L&15 ; halves e  -> K=k0+(L>=16)*16+e   (two b128 loads)
// B-frag  : lane L col n=L&15 ; halves e  -> K=k0+(L>=16)*16+e (contiguous in Whh row n)
// C layout: vgpr r, lane L -> M=r+(L>=16)*8, N=L&15
__device__ __forceinline__ void gate_gemm(
    const _Float16* __restrict__ lds_h,  // LDS: (64, LDS_ROW_H) f16 hidden panel
    const _Float16* __restrict__ whh16,  // (2,2048,512) f16
    const _Float16* __restrict__ wih16,  // (2,2048,128) f16 or nullptr
    const _Float16* __restrict__ sig16,  // (64,128,128) f16 signal
    float* __restrict__ gbuf,            // (2,64,2048)
    int d, int nblk, int wave, int lane, int tk /* <0: no input term */)
{
    const int l15 = lane & 15;
    const int hi  = lane >> 4;                    // 0 or 1
    const int mt  = wave & 3;                     // M tile (batch rows mt*16..)
    const int m   = mt * 16 + l15;                // batch row for A-frag
    const int n0  = nblk * 64 + (wave >> 2) * 32 + l15;
    const int n1  = n0 + 16;

    const _Float16* hrow = lds_h + (size_t)m * LDS_ROW_H;
    const _Float16* w0   = whh16 + ((size_t)d * FH_ + n0) * H_;
    const _Float16* w1   = whh16 + ((size_t)d * FH_ + n1) * H_;

    v8f acc0 = {}; v8f acc1 = {};

#pragma unroll 4
    for (int kk = 0; kk < 16; ++kk) {
        const int k0 = kk * 32;
        v8h alo = *(const v8h*)(hrow + k0 + hi * 16);
        v8h ahi = *(const v8h*)(hrow + k0 + hi * 16 + 8);
        v16h a;
#pragma unroll
        for (int i = 0; i < 8; ++i) { a[i] = alo[i]; a[i + 8] = ahi[i]; }
        v16h b0 = *(const v16h*)(w0 + k0 + hi * 16);
        v16h b1 = *(const v16h*)(w1 + k0 + hi * 16);
        acc0 = __builtin_amdgcn_wmma_f32_16x16x32_f16(false, a, false, b0, (short)0, acc0, false, false);
        acc1 = __builtin_amdgcn_wmma_f32_16x16x32_f16(false, a, false, b1, (short)0, acc1, false, false);
    }

    if (tk >= 0) {  // fold x_t @ Wih^T into the same accumulators (K = 512..639)
        const _Float16* xrow = sig16 + ((size_t)m * T_ + tk) * I_;
        const _Float16* wi0  = wih16 + ((size_t)d * FH_ + n0) * I_;
        const _Float16* wi1  = wih16 + ((size_t)d * FH_ + n1) * I_;
#pragma unroll
        for (int kk = 0; kk < 4; ++kk) {
            const int k0 = kk * 32;
            v8h alo = *(const v8h*)(xrow + k0 + hi * 16);
            v8h ahi = *(const v8h*)(xrow + k0 + hi * 16 + 8);
            v16h a;
#pragma unroll
            for (int i = 0; i < 8; ++i) { a[i] = alo[i]; a[i + 8] = ahi[i]; }
            v16h b0 = *(const v16h*)(wi0 + k0 + hi * 16);
            v16h b1 = *(const v16h*)(wi1 + k0 + hi * 16);
            acc0 = __builtin_amdgcn_wmma_f32_16x16x32_f16(false, a, false, b0, (short)0, acc0, false, false);
            acc1 = __builtin_amdgcn_wmma_f32_16x16x32_f16(false, a, false, b1, (short)0, acc1, false, false);
        }
    }

    float* g = gbuf + (size_t)d * 64 * FH_;
    const int mr = mt * 16 + hi * 8;
#pragma unroll
    for (int r = 0; r < 8; ++r) {
        g[(size_t)(mr + r) * FH_ + n0] = acc0[r];
        g[(size_t)(mr + r) * FH_ + n1] = acc1[r];
    }
}

// Elementwise LSTM cell over (2,64,512); 4 elements per thread (64 WGs * 256 thr).
__device__ __forceinline__ void cell_phase(
    const float* __restrict__ gbuf, const float* __restrict__ bsum,
    float* __restrict__ cbuf, _Float16* __restrict__ h16,
    const float* __restrict__ decWih /* nullptr for encoder */,
    const int* __restrict__ tok, int gid)
{
#pragma unroll
    for (int it = 0; it < 4; ++it) {
        int e = it * (NWG * 256) + gid;   // 0..65535
        int d = e >> 15;
        int rj = e & 32767;
        int b = rj >> 9;
        int j = rj & 511;
        const float* gr = gbuf + ((size_t)(d * 64 + b)) * FH_;
        const float* bs = bsum + (size_t)d * FH_;
        float gi = gr[j]            + bs[j];
        float gf = gr[H_ + j]       + bs[H_ + j];
        float gg = gr[2 * H_ + j]   + bs[2 * H_ + j];
        float go = gr[3 * H_ + j]   + bs[3 * H_ + j];
        if (decWih) {  // x = relu(one_hot(tok)) -> column gather of Wih
            int tkb = tok[b];
            const float* wd = decWih + (size_t)d * FH_ * V_;
            gi += wd[(size_t)j * V_ + tkb];
            gf += wd[(size_t)(H_ + j) * V_ + tkb];
            gg += wd[(size_t)(2 * H_ + j) * V_ + tkb];
            go += wd[(size_t)(3 * H_ + j) * V_ + tkb];
        }
        size_t hidx = (size_t)(d * 64 + b) * H_ + j;
        float c  = cbuf[hidx];
        float cn = sigm(gf) * c + sigm(gi) * tanhf(gg);
        float hn = sigm(go) * tanhf(cn);
        cbuf[hidx] = cn;
        h16[hidx]  = (_Float16)hn;
    }
}

// ---------------- init: f32->f16 weight/signal conversion, bias sums, state zeroing ----------------
__global__ __launch_bounds__(256) void lstm_init_kernel(
    const float* __restrict__ signal, const float* __restrict__ tgt,
    const float* __restrict__ encWih, const float* __restrict__ encWhh,
    const float* __restrict__ encBih, const float* __restrict__ encBhh,
    const float* __restrict__ decWhh, const float* __restrict__ decBih, const float* __restrict__ decBhh,
    _Float16* __restrict__ encWhh16, _Float16* __restrict__ decWhh16,
    _Float16* __restrict__ encWih16, _Float16* __restrict__ sig16,
    float* __restrict__ bsE, float* __restrict__ bsD,
    _Float16* __restrict__ h16, float* __restrict__ cbuf,
    int* __restrict__ tok, unsigned* __restrict__ cnt, unsigned* __restrict__ gen)
{
    const size_t i0 = (size_t)blockIdx.x * blockDim.x + threadIdx.x;
    const size_t st = (size_t)gridDim.x * blockDim.x;
    for (size_t i = i0; i < (size_t)2 * FH_ * H_; i += st) {
        encWhh16[i] = (_Float16)encWhh[i];
        decWhh16[i] = (_Float16)decWhh[i];
    }
    for (size_t i = i0; i < (size_t)B_ * T_ * I_; i += st) sig16[i] = (_Float16)signal[i];
    for (size_t i = i0; i < (size_t)2 * FH_ * I_;  i += st) encWih16[i] = (_Float16)encWih[i];
    for (size_t i = i0; i < (size_t)2 * B_ * H_;   i += st) { h16[i] = (_Float16)0.0f; cbuf[i] = 0.0f; }
    for (size_t i = i0; i < (size_t)2 * FH_;       i += st) { bsE[i] = encBih[i] + encBhh[i]; bsD[i] = decBih[i] + decBhh[i]; }
    for (size_t i = i0; i < (size_t)B_;            i += st) {
        const float* row = tgt + i * (L_ * V_);  // target_onehot[b, 0, :]
        int best = 0; float bv = row[0];
        for (int v = 1; v < V_; ++v) { if (row[v] > bv) { bv = row[v]; best = v; } }
        tok[i] = best;
    }
    if (i0 == 0) { *cnt = 0u; *gen = 0u; }
}

// ---------------- persistent seq2seq kernel ----------------
__global__ __launch_bounds__(256, 1) void lstm_seq2seq_kernel(
    const float* __restrict__ decWih, const float* __restrict__ Wout, const float* __restrict__ bout,
    const _Float16* __restrict__ encWhh16, const _Float16* __restrict__ decWhh16,
    const _Float16* __restrict__ encWih16, const _Float16* __restrict__ sig16,
    const float* __restrict__ bsE, const float* __restrict__ bsD,
    _Float16* __restrict__ h16, float* __restrict__ cbuf, float* __restrict__ gbuf,
    float* __restrict__ lbuf, int* __restrict__ tok,
    unsigned* __restrict__ cnt, unsigned* __restrict__ gen,
    float* __restrict__ out)
{
    extern __shared__ _Float16 lds_h[];   // (64, LDS_ROW_H) padded hidden panel

    const int wg   = blockIdx.x;       // 0..63
    const int tid  = threadIdx.x;      // 0..255 (8 waves of 32)
    const int wave = tid >> 5;
    const int lane = tid & 31;
    const int d    = wg >> 5;          // direction for GEMM phase
    const int nblk = wg & 31;          // 64-column block of the 2048 gate dim
    const int gid  = wg * 256 + tid;   // 0..16383
    unsigned rnd = 0;

    // ---- TDM descriptor for this direction's h panel (built once, reused every step) ----
    // Group0: count=1 | lds_addr=0 | global_addr (57b) | type=2
    unsigned long long ga = (unsigned long long)(uintptr_t)(h16 + (size_t)d * 64 * H_);
    v4u g0 = { 1u, 0u,
               (unsigned)(ga & 0xFFFFFFFFu),
               (unsigned)((ga >> 32) & 0x01FFFFFFu) | 0x80000000u };
    // Group1: data_size=1(2B), pad_enable=1, pad_interval=7(256 dwords), pad_amount=3(4 dwords),
    //         tensor_dim0=512, tensor_dim1=64, tile_dim0=512, tile_dim1=64, dim0_stride=512
    v8i g1;
    g1[0] = (1 << 16) | (1 << 20) | (7 << 22) | (3 << 25);
    g1[1] = (int)(512u << 16);   // tensor_dim0[15:0] in bits[31:16]
    g1[2] = (int)(64u << 16);    // tensor_dim0 hi=0 ; tensor_dim1[15:0]=64
    g1[3] = (int)(512u << 16);   // tensor_dim1 hi=0 ; tile_dim0=512
    g1[4] = 64;                  // tile_dim1=64 ; tile_dim2=0
    g1[5] = 512;                 // tensor_dim0_stride[31:0]
    g1[6] = 0;
    g1[7] = 0;

    // ---------------- encoder: 128 recurrent steps, both directions in parallel ----------------
    for (int t = 0; t < T_; ++t) {
        const int tk = (d == 0) ? t : (T_ - 1 - t);
        stage_h(g0, g1);                                   // TDM: h16[d] -> LDS (padded)
        gate_gemm(lds_h, encWhh16, encWih16, sig16, gbuf, d, nblk, wave, lane, tk);
        grid_sync(cnt, gen, ++rnd);
        cell_phase(gbuf, bsE, cbuf, h16, nullptr, tok, gid);
        grid_sync(cnt, gen, ++rnd);
    }

    // ---------------- decoder: 64 steps with argmax feedback ----------------
    for (int t = 0; t < L_; ++t) {
        stage_h(g0, g1);
        gate_gemm(lds_h, decWhh16, nullptr, nullptr, gbuf, d, nblk, wave, lane, -1);
        grid_sync(cnt, gen, ++rnd);
        cell_phase(gbuf, bsD, cbuf, h16, decWih, tok, gid);
        grid_sync(cnt, gen, ++rnd);

        // logits[b,v] = b_out[v] + [hf|hb] . W_out[v,:]   (2048 lanes, K=1024)
        if (gid < B_ * V_) {
            int b = gid >> 5, v = gid & 31;
            float s = bout[v];
            const _Float16* hf = h16 + (size_t)b * H_;
            const _Float16* hb = h16 + (size_t)(B_ + b) * H_;
            const float* wr = Wout + (size_t)v * (2 * H_);
            for (int j = 0; j < H_; ++j)
                s += (float)hf[j] * wr[j] + (float)hb[j] * wr[H_ + j];
            lbuf[b * V_ + v] = s;
        }
        grid_sync(cnt, gen, ++rnd);

        // per-row argmax + log_softmax: one wave32 per batch row (V==32 lanes exactly)
        if (wg < 8) {
            int b = wg * 8 + wave;
            float x = lbuf[b * V_ + lane];
            float bv = x; int bi = lane;
#pragma unroll
            for (int off = 16; off > 0; off >>= 1) {
                float ov = __shfl_xor(bv, off, 32);
                int   oi = __shfl_xor(bi, off, 32);
                if (ov > bv || (ov == bv && oi < bi)) { bv = ov; bi = oi; }
            }
            float s = __expf(x - bv);
#pragma unroll
            for (int off = 16; off > 0; off >>= 1) s += __shfl_xor(s, off, 32);
            out[((size_t)b * L_ + t) * V_ + lane] = x - bv - __logf(s);
            if (lane == 0) tok[b] = bi;
        }
        grid_sync(cnt, gen, ++rnd);
    }
}

extern "C" void kernel_launch(void* const* d_in, const int* in_sizes, int n_in,
                              void* d_out, int out_size, void* d_ws, size_t ws_size,
                              hipStream_t stream)
{
    (void)in_sizes; (void)n_in; (void)out_size;
    const float* signal = (const float*)d_in[0];
    const float* tgt    = (const float*)d_in[1];
    const float* encWih = (const float*)d_in[2];
    const float* encWhh = (const float*)d_in[3];
    const float* encBih = (const float*)d_in[4];
    const float* encBhh = (const float*)d_in[5];
    const float* decWih = (const float*)d_in[6];
    const float* decWhh = (const float*)d_in[7];
    const float* decBih = (const float*)d_in[8];
    const float* decBhh = (const float*)d_in[9];
    const float* Wout   = (const float*)d_in[10];
    const float* bout   = (const float*)d_in[11];

    char* ws = (char*)d_ws;
    if (ws_size < WS_NEED) return;  // insufficient scratch; cannot proceed safely

    unsigned*  cnt      = (unsigned*)(ws + OFF_CNT);
    unsigned*  gen      = (unsigned*)(ws + OFF_GEN);
    int*       tok      = (int*)     (ws + OFF_TOK);
    _Float16*  h16      = (_Float16*)(ws + OFF_H16);
    float*     cbuf     = (float*)   (ws + OFF_CB);
    float*     gbuf     = (float*)   (ws + OFF_GB);
    float*     lbuf     = (float*)   (ws + OFF_LB);
    float*     bsE      = (float*)   (ws + OFF_BSE);
    float*     bsD      = (float*)   (ws + OFF_BSD);
    _Float16*  encWhh16 = (_Float16*)(ws + OFF_EWHH);
    _Float16*  decWhh16 = (_Float16*)(ws + OFF_DWHH);
    _Float16*  encWih16 = (_Float16*)(ws + OFF_EWIH);
    _Float16*  sig16    = (_Float16*)(ws + OFF_SIG);

    lstm_init_kernel<<<512, 256, 0, stream>>>(
        signal, tgt, encWih, encWhh, encBih, encBhh, decWhh, decBih, decBhh,
        encWhh16, decWhh16, encWih16, sig16, bsE, bsD, h16, cbuf, tok, cnt, gen);

    lstm_seq2seq_kernel<<<NWG, 256, LDS_BYTES, stream>>>(
        decWih, Wout, bout, encWhh16, decWhh16, encWih16, sig16, bsE, bsD,
        h16, cbuf, gbuf, lbuf, tok, cnt, gen, (float*)d_out);
}
